// GraphConvNet_7052336300582
// MI455X (gfx1250) — matrix operand order repaired and checked by reading, not compiled, not run on hardware
//
#include <hip/hip_runtime.h>
#include <hip/hip_bf16.h>

// GraphConvNet for MI455X (gfx1250, wave32, WMMA + TDM).
// Pipeline:
//  1) zero agg (f32, ws)
//  2) edge scatter-add: wave-per-edge, coalesced float4 gather + f32 atomics
//  3) pack [agg | x] -> bf16 Acat[N,256]; pack [W_rel | W_root] -> bf16 Wcat[128,256];
//     convert W1,W2 -> bf16
//  4) GEMM1: h  = Acat @ Wcat^T + b_rel   (K=256, NC=128)  v_wmma_f32_16x16x32_bf16
//  5) GEMM2: h1 = relu(h @ W1^T + b1)     (K=128, NC=256)
//  6) GEMM3: h2 = relu(h1 @ W2^T + b2)    (K=256, NC=256)
//  7) head:  out = sigmoid(h2 . W3 + b3)  wave-per-node reduction
// GEMM weights are staged into LDS once per block with the Tensor Data Mover
// (tensor_load_to_lds + s_wait_tensorcnt), with TDM padding for a 16B row skew
// so B-fragment ds_load_b128 reads are bank-conflict-free.

#define NNODES 10000
#define DIM    128
#define HID    256

typedef __attribute__((ext_vector_type(16))) __bf16        bf16x16;
typedef __attribute__((ext_vector_type(8)))  float         f32x8;
typedef __attribute__((ext_vector_type(4)))  unsigned int  u32x4;
typedef __attribute__((ext_vector_type(4)))  unsigned int  uint32x4;
typedef __attribute__((ext_vector_type(8)))  int           int32x8;
typedef __attribute__((ext_vector_type(4)))  int           int32x4;

union Frag16 {
    bf16x16 v;
    u32x4   u[2];
};

// ---------------------------------------------------------------- utilities

__global__ __launch_bounds__(256) void gcn_zero_f32(float* __restrict__ p, int n) {
    int i = blockIdx.x * 256 + threadIdx.x;
    if (i < n) p[i] = 0.0f;
}

__global__ __launch_bounds__(256) void gcn_cvt_f32_bf16(const float* __restrict__ in,
                                                        __bf16* __restrict__ out, int n) {
    int i = blockIdx.x * 256 + threadIdx.x;
    if (i < n) out[i] = (__bf16)in[i];
}

// Acat[i, 128 + k] = x[i, k]
__global__ __launch_bounds__(256) void gcn_pack_x(const float* __restrict__ x,
                                                  __bf16* __restrict__ Acat, int n) {
    int idx = blockIdx.x * 256 + threadIdx.x;
    if (idx < n) {
        int i = idx >> 7;          // / DIM
        int k = idx & (DIM - 1);
        Acat[(size_t)i * (2 * DIM) + DIM + k] = (__bf16)x[idx];
    }
}

// Acat[i, k] = agg[i, k]
__global__ __launch_bounds__(256) void gcn_pack_agg(const float* __restrict__ agg,
                                                    __bf16* __restrict__ Acat, int n) {
    int idx = blockIdx.x * 256 + threadIdx.x;
    if (idx < n) {
        int i = idx >> 7;
        int k = idx & (DIM - 1);
        Acat[(size_t)i * (2 * DIM) + k] = (__bf16)agg[idx];
    }
}

// Wcat[j, k] = (k < 128) ? W_rel[j, k] : W_root[j, k-128]   (j in [0,128), k in [0,256))
__global__ __launch_bounds__(256) void gcn_pack_wcat(const float* __restrict__ Wrel,
                                                     const float* __restrict__ Wroot,
                                                     __bf16* __restrict__ Wcat) {
    int idx = blockIdx.x * 256 + threadIdx.x;   // 128*256 elements
    if (idx < DIM * 2 * DIM) {
        int j = idx >> 8;           // / 256
        int k = idx & 255;
        float v = (k < DIM) ? Wrel[(size_t)j * DIM + k]
                            : Wroot[(size_t)j * DIM + (k - DIM)];
        Wcat[idx] = (__bf16)v;
    }
}

// ------------------------------------------------------ edge scatter-add
// One wave per edge; lane c owns floats [4c, 4c+4). Gather x[src] row (512B
// coalesced per wave), atomically add into agg[dst] row (contiguous f32 atomics).
__global__ __launch_bounds__(256) void gcn_scatter(const float* __restrict__ x,
                                                   const int* __restrict__ ei,
                                                   float* __restrict__ agg, int E) {
    int gtid = blockIdx.x * 256 + threadIdx.x;
    int e = gtid >> 5;
    if (e >= E) return;
    int c = gtid & 31;
    int s = ei[e];
    int d = ei[E + e];
    const float4 v = *(const float4*)(x + (size_t)s * DIM + 4 * c);
    float* dp = agg + (size_t)d * DIM + 4 * c;
    atomicAdd(dp + 0, v.x);
    atomicAdd(dp + 1, v.y);
    atomicAdd(dp + 2, v.z);
    atomicAdd(dp + 3, v.w);
}

// ------------------------------------------------------ WMMA GEMM with TDM-staged weights
// out[M, NC] = act(A[M, K] @ W[NC, K]^T + bias)
// Wave tile: 16 rows x 64 cols (4 sub-tiles of 16x16), K stepped by 32.
// Fragment layouts per CDNA5 ISA 7.12.2 (wave32):
//   A 16x32 bf16: lane L (m = L%16, half = L/16) holds K = {8*half..+7} and
//                 {16+8*half..+7}  -> two contiguous b128 global loads.
//   B 32x16 bf16: lane L (n = L%16, half = L/16) holds K = {8*half..+7} and
//                 {16+8*half..+7} of column n == row n of W  -> two b128 LDS loads.
//   C/D f32:     VGPR r, lane L -> (m = r + 8*(L/16), n = L%16).
// W is staged [NC, K] -> LDS with a 16B pad per row (row stride K+8 elements,
// i.e. 132/68 DWORDs == 4 mod 64 banks: conflict-free column access).
template <int K, int NC, bool RELU>
__global__ __launch_bounds__(256) void gcn_gemm_wmma(const __bf16* __restrict__ A,
                                                     const __bf16* __restrict__ W,
                                                     const float* __restrict__ bias,
                                                     __bf16* __restrict__ out, int M) {
    constexpr int SROW = K + 8;                    // padded LDS row stride (elements)
    __shared__ __align__(16) __bf16 smemW[NC * SROW];

#if __has_builtin(__builtin_amdgcn_tensor_load_to_lds)
    // ---- stage W[NC,K] into LDS via the Tensor Data Mover (wave 0 issues) ----
    if (threadIdx.x < 32) {
        constexpr unsigned padInterval = (K == 256) ? 6u : 5u;  // 2^(i+1) dw == K/2
        unsigned long long ga = (unsigned long long)W;
        unsigned lds = (unsigned)(unsigned long long)(void*)smemW;
        uint32x4 g0;
        g0.x = 1u;                                   // count=1 (valid descriptor)
        g0.y = lds;                                  // lds_addr
        g0.z = (unsigned)(ga & 0xffffffffu);         // global_addr[31:0]
        g0.w = (unsigned)((ga >> 32) & 0x01ffffffu)  // global_addr[56:32]
             | (2u << 30);                           // D# type = 2 ("image")
        int32x8 g1;
        g1[0] = (int)((1u << 16)                     // data_size = 2 bytes
                    | (1u << 20)                     // pad_enable
                    | (padInterval << 22)            // pad_interval = row length
                    | (3u << 25));                   // pad_amount = 4 DWORDs (16B)
        g1[1] = (int)((K & 0xffff) << 16);           // tensor_dim0[15:0]
        g1[2] = (int)(((unsigned)K >> 16) | ((NC & 0xffff) << 16)); // dim0 hi | dim1 lo
        g1[3] = (int)(((unsigned)NC >> 16) | ((K & 0xffff) << 16)); // dim1 hi | tile_dim0=K
        g1[4] = (int)(NC & 0xffff);                  // tile_dim1=NC, tile_dim2=0
        g1[5] = (int)K;                              // tensor_dim0_stride = K
        g1[6] = 0;
        g1[7] = 0;
        int32x4 gz = {0, 0, 0, 0};
#if __clang_major__ >= 23
        int32x8 gz8 = {0, 0, 0, 0, 0, 0, 0, 0};
        __builtin_amdgcn_tensor_load_to_lds(g0, g1, gz, gz, gz8, 0);
#else
        __builtin_amdgcn_tensor_load_to_lds(g0, g1, gz, gz, 0);
#endif
        __builtin_amdgcn_s_wait_tensorcnt(0);
    }
#else
    // fallback: cooperative copy with the same padded layout
    for (int i = threadIdx.x; i < NC * K; i += 256) {
        smemW[(i / K) * SROW + (i % K)] = W[i];
    }
#endif
    __syncthreads();

    const int lane   = threadIdx.x & 31;
    const int waveId = blockIdx.x * 8 + (threadIdx.x >> 5);
    const int colTiles = NC / 64;
    const int rowTiles = M / 16;            // M == 10000 -> 625 exact tiles
    if (waveId >= rowTiles * colTiles) return;   // wave-uniform exit; EXEC stays full

    const int rowBase = (waveId / colTiles) * 16;
    const int colBase = (waveId % colTiles) * 64;
    const int m16  = lane & 15;
    const int half = lane >> 4;

    f32x8 acc[4];
    acc[0] = f32x8{};
    acc[1] = f32x8{};
    acc[2] = f32x8{};
    acc[3] = f32x8{};

    const __bf16* Arow = A + (size_t)(rowBase + m16) * K + 8 * half;

#pragma unroll
    for (int k = 0; k < K; k += 32) {
        if (k + 64 < K) __builtin_prefetch(Arow + k + 64, 0, 1);  // global_prefetch_b8
        Frag16 a;
        a.u[0] = *(const u32x4*)(Arow + k);        // K = k + 8*half + [0..7]
        a.u[1] = *(const u32x4*)(Arow + k + 16);   // K = k + 16 + 8*half + [0..7]
#pragma unroll
        for (int s = 0; s < 4; ++s) {
            const __bf16* Bp = smemW + (colBase + s * 16 + m16) * SROW + k + 8 * half;
            Frag16 b;
            b.u[0] = *(const u32x4*)(Bp);          // ds_load_b128: K = k+8*half+[0..7]
            b.u[1] = *(const u32x4*)(Bp + 16);     // ds_load_b128: K = k+16+8*half+[0..7]
            acc[s] = __builtin_amdgcn_wmma_f32_16x16x32_bf16(
                false, a.v, false, b.v, (short)0, acc[s], false, false);
        }
    }

#pragma unroll
    for (int s = 0; s < 4; ++s) {
        const int n = colBase + s * 16 + m16;
        const float bv = bias[n];
        f32x8 c = acc[s];
#pragma unroll
        for (int r = 0; r < 8; ++r) {
            float v = c[r] + bv;
            if (RELU) v = v > 0.0f ? v : 0.0f;
            out[(size_t)(rowBase + r + 8 * half) * NC + n] = (__bf16)v;
        }
    }
}

// ------------------------------------------------------ head: dot(256) + sigmoid
__global__ __launch_bounds__(256) void gcn_head(const __bf16* __restrict__ h2,
                                                const float* __restrict__ W3,
                                                const float* __restrict__ b3,
                                                float* __restrict__ out, int M) {
    int gtid = blockIdx.x * 256 + threadIdx.x;
    int i = gtid >> 5;
    if (i >= M) return;
    int lane = gtid & 31;
    float s = 0.0f;
#pragma unroll
    for (int j = 0; j < HID / 32; ++j) {
        int k = lane + 32 * j;
        s += (float)h2[(size_t)i * HID + k] * W3[k];
    }
#pragma unroll
    for (int off = 16; off > 0; off >>= 1) s += __shfl_xor(s, off, 32);
    if (lane == 0) out[i] = 1.0f / (1.0f + __expf(-(s + b3[0])));
}

// ------------------------------------------------------ launcher

extern "C" void kernel_launch(void* const* d_in, const int* in_sizes, int n_in,
                              void* d_out, int out_size, void* d_ws, size_t ws_size,
                              hipStream_t stream) {
    const float* x      = (const float*)d_in[0];
    const int*   ei     = (const int*)d_in[1];     // edge_index [2, E] (int32)
    const float* W_rel  = (const float*)d_in[2];
    const float* b_rel  = (const float*)d_in[3];
    const float* W_root = (const float*)d_in[4];
    const float* W1     = (const float*)d_in[5];
    const float* b1     = (const float*)d_in[6];
    const float* W2     = (const float*)d_in[7];
    const float* b2     = (const float*)d_in[8];
    const float* W3     = (const float*)d_in[9];
    const float* b3     = (const float*)d_in[10];
    float* out = (float*)d_out;

    const int N = in_sizes[0] / DIM;        // 10000
    const int E = in_sizes[1] / 2;          // 640000

    // workspace carve-out (256B aligned)
    char* wsb = (char*)d_ws;
    size_t o = 0;
    auto take = [&](size_t bytes) -> char* {
        char* p = wsb + o;
        o += (bytes + 255) & ~(size_t)255;
        return p;
    };
    float*  agg  = (float*) take((size_t)N * DIM * 4);          // f32 aggregation
    __bf16* Acat = (__bf16*)take((size_t)N * 2 * DIM * 2);      // [N, 256] bf16
    __bf16* Wcat = (__bf16*)take((size_t)DIM * 2 * DIM * 2);    // [128, 256] bf16
    __bf16* W1h  = (__bf16*)take((size_t)HID * DIM * 2);        // [256, 128] bf16
    __bf16* W2h  = (__bf16*)take((size_t)HID * HID * 2);        // [256, 256] bf16
    __bf16* h    = (__bf16*)take((size_t)N * DIM * 2);          // [N, 128] bf16
    __bf16* h1   = (__bf16*)take((size_t)N * HID * 2);          // [N, 256] bf16
    __bf16* h2   = (__bf16*)take((size_t)N * HID * 2);          // [N, 256] bf16
    (void)ws_size;

    auto blocks = [](long long n) { return (unsigned)((n + 255) / 256); };

    // 1) zero aggregation buffer
    gcn_zero_f32<<<blocks((long long)N * DIM), 256, 0, stream>>>(agg, N * DIM);

    // 2) edge scatter-add (wave-per-edge)
    gcn_scatter<<<blocks((long long)E * 32), 256, 0, stream>>>(x, ei, agg, E);

    // 3) packing / precision conversion
    gcn_pack_x  <<<blocks((long long)N * DIM), 256, 0, stream>>>(x,   Acat, N * DIM);
    gcn_pack_agg<<<blocks((long long)N * DIM), 256, 0, stream>>>(agg, Acat, N * DIM);
    gcn_pack_wcat<<<blocks(DIM * 2 * DIM), 256, 0, stream>>>(W_rel, W_root, Wcat);
    gcn_cvt_f32_bf16<<<blocks(HID * DIM), 256, 0, stream>>>(W1, W1h, HID * DIM);
    gcn_cvt_f32_bf16<<<blocks(HID * HID), 256, 0, stream>>>(W2, W2h, HID * HID);

    // 4-6) WMMA GEMM chain (weights TDM-staged to LDS per block)
    {
        unsigned waves = (unsigned)(N / 16) * (DIM / 64);   // 625 * 2
        gcn_gemm_wmma<2 * DIM, DIM, false>
            <<<(waves + 7) / 8, 256, 0, stream>>>(Acat, Wcat, b_rel, h, N);
    }
    {
        unsigned waves = (unsigned)(N / 16) * (HID / 64);   // 625 * 4
        gcn_gemm_wmma<DIM, HID, true>
            <<<(waves + 7) / 8, 256, 0, stream>>>(h, W1h, b1, h1, N);
    }
    {
        unsigned waves = (unsigned)(N / 16) * (HID / 64);   // 625 * 4
        gcn_gemm_wmma<HID, HID, true>
            <<<(waves + 7) / 8, 256, 0, stream>>>(h1, W2h, b2, h2, N);
    }

    // 7) head
    gcn_head<<<blocks((long long)N * 32), 256, 0, stream>>>(h2, W3, b3, out, N);
}
